// RNNModel_14834817040872
// MI455X (gfx1250) — matrix operand order repaired and checked by reading, not compiled, not run on hardware
//
#include <hip/hip_runtime.h>

// ---------------------------------------------------------------------------
// RNNoise-style fused persistent GRU chain for gfx1250 (MI455X).
//   - dense layer precomputed chip-wide (input-only -> off the serial path)
//   - one workgroup per 16-row batch tile, 8 wave32, full T loop in-kernel
//   - gx/gh GEMMs of each GRU fused into one balanced WMMA job list
//   - biases folded into the K-padding rows of the packed B-fragments
//     (A-pad column pinned to 1.0), so WMMA outputs arrive pre-biased
//   - fast native sigmoid/tanh: v_exp_f32 + v_rcp_f32 (no IEEE div chains)
// ---------------------------------------------------------------------------

typedef __attribute__((ext_vector_type(16))) _Float16 v16h;
typedef __attribute__((ext_vector_type(8)))  float    v8f;
typedef __attribute__((ext_vector_type(4)))  unsigned int v4u;

union F16Frag { v16h h; v4u u[2]; };

#define NWAVES 8
constexpr int B_ = 256, T_ = 2000, F_ = 42;

// Fragment table (fragment = 16x16x32 B tile = 512 f16 = 1KB).
// matrix          NT  KT  N    K    bias@k=K  fragBase
// dense_w          2   2  24   42   yes       0
// vad_w_ih         5   1  72   24   yes       4
// vad_w_hh         5   1  72   24   yes       9
// noise_w_ih       9   3  144  90   yes       14
// noise_w_hh       9   2  144  48   yes       41
// den_w_ih        18   4  288  114  yes       59
// den_w_hh        18   3  288  96   no (K%32==0)  131
// out_w            2   3  22   96   no        185    total = 191 frags
constexpr int FR_DENSE = 0, FR_VIH = 4, FR_VHH = 9, FR_NIH = 14, FR_NHH = 41,
              FR_DIH = 59, FR_DHH = 131, FR_OUT = 185, FR_TOTAL = 191;

constexpr size_t TMP_OFF   = 262144;                    // tmp buffer offset in d_ws
constexpr size_t TMP_BYTES = (size_t)B_ * T_ * 24 * 2;  // 24.6 MB f16

// ---------------------------------------------------------------------------
// Weight packer: W is [N,K] row-major f32; B[k][n] = W[n][k].
// Fragment element (lane, e): n_local = lane&15, k_local = (e>>3)*16 + (lane>>4)*8 + (e&7).
// Row k==K carries the bias (paired with a 1.0 pinned in the A pad column);
// remaining padding is zero.
// ---------------------------------------------------------------------------
__global__ void pack_weights_f16(const float* __restrict__ dw,  const float* __restrict__ db,
                                 const float* __restrict__ vih, const float* __restrict__ vbi,
                                 const float* __restrict__ vhh, const float* __restrict__ vbh,
                                 const float* __restrict__ nih, const float* __restrict__ nbi,
                                 const float* __restrict__ nhh, const float* __restrict__ nbh,
                                 const float* __restrict__ dih, const float* __restrict__ dbi,
                                 const float* __restrict__ dhh, const float* __restrict__ ow,
                                 _Float16* __restrict__ dst) {
  int gid = blockIdx.x * blockDim.x + threadIdx.x;
  if (gid >= FR_TOTAL * 512) return;
  int frag = gid >> 9, e512 = gid & 511;

  const float* W; const float* Bs; int base, KT, N, K;
  if      (frag < FR_VIH) { W = dw;  Bs = db;      base = FR_DENSE; KT = 2; N = 24;  K = 42;  }
  else if (frag < FR_VHH) { W = vih; Bs = vbi;     base = FR_VIH;   KT = 1; N = 72;  K = 24;  }
  else if (frag < FR_NIH) { W = vhh; Bs = vbh;     base = FR_VHH;   KT = 1; N = 72;  K = 24;  }
  else if (frag < FR_NHH) { W = nih; Bs = nbi;     base = FR_NIH;   KT = 3; N = 144; K = 90;  }
  else if (frag < FR_DIH) { W = nhh; Bs = nbh;     base = FR_NHH;   KT = 2; N = 144; K = 48;  }
  else if (frag < FR_DHH) { W = dih; Bs = dbi;     base = FR_DIH;   KT = 4; N = 288; K = 114; }
  else if (frag < FR_OUT) { W = dhh; Bs = nullptr; base = FR_DHH;   KT = 3; N = 288; K = 96;  }
  else                    { W = ow;  Bs = nullptr; base = FR_OUT;   KT = 3; N = 22;  K = 96;  }

  int f  = frag - base;
  int nt = f / KT, kt = f - nt * KT;
  int lane = e512 >> 4, e = e512 & 15;
  int n = nt * 16 + (lane & 15);
  int k = kt * 32 + ((e >> 3) << 4) + ((lane >> 4) << 3) + (e & 7);
  float v = 0.f;
  if (n < N) {
    if (k < K)                  v = W[n * K + k];
    else if (k == K && Bs)      v = Bs[n];       // fused bias row
  }
  dst[gid] = (_Float16)v;
}

// ---------------------------------------------------------------------------
// Chip-wide precompute of tmp = tanh(x @ dense_w.T + b): input-only, runs
// once at full parallelism instead of 2000x on the serial path.
// ---------------------------------------------------------------------------
__global__ __launch_bounds__(256) void dense_tmp_precompute(
    const float* __restrict__ x, const float* __restrict__ dw,
    const float* __restrict__ db, _Float16* __restrict__ tmp) {
  size_t gid = (size_t)blockIdx.x * blockDim.x + threadIdx.x;
  if (gid >= (size_t)B_ * T_ * 24) return;
  int o = (int)(gid % 24);
  size_t bt = gid / 24;
  const float* xr = x + bt * F_;
  const float* wr = dw + o * F_;
  float s = db[o];
#pragma unroll 6
  for (int f = 0; f < F_; ++f) s += xr[f] * wr[f];
  tmp[gid] = (_Float16)tanhf(s);
}

// Fast native activations: v_exp_f32 + v_rcp_f32 (no IEEE divide chain).
__device__ __forceinline__ float fast_rcp(float x)  { return __builtin_amdgcn_rcpf(x); }
__device__ __forceinline__ float sigmoidf_(float x) { return fast_rcp(1.f + __expf(-x)); }
__device__ __forceinline__ float tanhf_(float x)    { return 1.f - 2.f * fast_rcp(1.f + __expf(2.f * x)); }

// ---------------------------------------------------------------------------
// One 16x16 output tile: A (16 x 32*KT) from row-major f16 LDS, B fragments
// from packed global (cache-resident), f32 accum, scatter into f32 LDS.
// ---------------------------------------------------------------------------
template <int KT>
__device__ __forceinline__ void wmma_tile(const _Float16* __restrict__ aRow,
                                          const _Float16* __restrict__ wp,
                                          float* __restrict__ g) {
  v8f acc = {0.f, 0.f, 0.f, 0.f, 0.f, 0.f, 0.f, 0.f};
#pragma unroll
  for (int kt = 0; kt < KT; ++kt) {
    F16Frag a, b;
    a.u[0] = *(const v4u*)(aRow + kt * 32);        // K = hi*8 .. +7
    a.u[1] = *(const v4u*)(aRow + kt * 32 + 16);   // K = 16+hi*8 .. +7
    const v4u* pb = (const v4u*)(wp + (kt << 9));
    b.u[0] = pb[0];
    b.u[1] = pb[1];
    acc = __builtin_amdgcn_wmma_f32_16x16x32_f16(false, a.h, false, b.h,
                                                 (short)0, acc, false, false);
  }
#pragma unroll
  for (int r = 0; r < 8; ++r) g[r * 288] = acc[r];
}

template <int NT, int KT>
__device__ __forceinline__ void gemm_one(const _Float16* __restrict__ aLDS, int aStride,
                                         const _Float16* __restrict__ wBase,
                                         float* __restrict__ gLDS, int wave, int lane) {
  const int mcol = lane & 15, hi = lane >> 4;
  constexpr int NIT = (NT + NWAVES - 1) / NWAVES;
#pragma unroll
  for (int jj = 0; jj < NIT; ++jj) {
    int nt = wave + jj * NWAVES;          // uniform per wave
    if (nt >= NT) break;
    wmma_tile<KT>(aLDS + mcol * aStride + hi * 8,
                  wBase + ((nt * KT) << 9) + (lane << 4),
                  gLDS + (hi * 8) * 288 + nt * 16 + mcol);
  }
}

// Fused gx+gh GEMMs of one GRU as a single balanced job list across waves.
template <int NTX, int KTX, int NTH, int KTH>
__device__ __forceinline__ void gemm_pair(
    const _Float16* __restrict__ aX, int asX, const _Float16* __restrict__ wX,
    float* __restrict__ gX,
    const _Float16* __restrict__ aH, int asH, const _Float16* __restrict__ wH,
    float* __restrict__ gH, int wave, int lane) {
  const int mcol = lane & 15, hi = lane >> 4;
  constexpr int NJOB = NTX + NTH;
  constexpr int NIT  = (NJOB + NWAVES - 1) / NWAVES;
#pragma unroll
  for (int jj = 0; jj < NIT; ++jj) {
    int job = wave + jj * NWAVES;         // uniform per wave
    if (job >= NJOB) break;
    if (job < NTX) {
      wmma_tile<KTX>(aX + mcol * asX + hi * 8,
                     wX + ((job * KTX) << 9) + (lane << 4),
                     gX + (hi * 8) * 288 + job * 16 + mcol);
    } else {
      int nt = job - NTX;
      wmma_tile<KTH>(aH + mcol * asH + hi * 8,
                     wH + ((nt * KTH) << 9) + (lane << 4),
                     gH + (hi * 8) * 288 + nt * 16 + mcol);
    }
  }
}

// ---------------------------------------------------------------------------
// Fused persistent kernel. grid = B/16, block = 256 (8 wave32).
// PRECOMP: dense layer comes from the precomputed tmp buffer (8 barriers/step)
// otherwise computed in-loop (fallback when workspace too small).
// ---------------------------------------------------------------------------
template <bool PRECOMP>
__global__ __launch_bounds__(256, 1) void rnnoise_fused(
    const float* __restrict__ x,            // [B,T,42]
    const _Float16* __restrict__ pw,        // packed weight fragments
    const _Float16* __restrict__ tmpg,      // [B,T,24] tanh(dense) if PRECOMP
    const float* __restrict__ vow, const float* __restrict__ vob,  // [24],[1]
    const float* __restrict__ dbh,          // den b_hh [288] (not fusable)
    const float* __restrict__ out_b,        // [22]
    float* __restrict__ dout, float* __restrict__ dvad) {
  __shared__ __align__(16) struct {
    _Float16 a_dense[16 * 64];   // x padded (42|1|0 -> 64)    (fallback only)
    _Float16 a_vad  [16 * 32];   // tmp (24|1|0 -> 32)
    _Float16 a_noise[16 * 96];   // [tmp | h_vad | x | 1 | 0]  (90|1 -> 96)
    _Float16 a_den  [16 * 128];  // [h_vad | relu(h_noise) | x | 1 | 0] (114|1->128)
    _Float16 a_out  [16 * 96];   // relu(h_den)
    _Float16 a_hv   [16 * 32];   // h_vad   (24|1|0 -> 32)
    _Float16 a_hn   [16 * 64];   // h_noise (48|1|0 -> 64)
    _Float16 a_hd   [16 * 96];   // h_den   (96, exact)
    float    gx[16 * 288];       // input-gate preacts incl. b_ih (stride 288)
    float    gh[16 * 288];       // hidden-gate preacts incl. b_hh (exc. den)
  } sm;

  const int tid  = threadIdx.x;
  const int wave = tid >> 5, lane = tid & 31;
  const int b0   = blockIdx.x * 16;

  // Stage timestep t's inputs into LDS (x regions, and tmp when PRECOMP).
  auto stage = [&](int t) {
    for (int i = tid; i < 16 * F_; i += 256) {
      int m = i / F_, f = i - m * F_;
      _Float16 h = (_Float16)x[((size_t)(b0 + m) * T_ + t) * F_ + f];
      if (!PRECOMP) sm.a_dense[m * 64 + f] = h;
      sm.a_noise[m * 96 + 48 + f]  = h;
      sm.a_den  [m * 128 + 72 + f] = h;
    }
    if (PRECOMP) {
      for (int i = tid; i < 16 * 24; i += 256) {
        int m = i / 24, j = i - m * 24;
        _Float16 h = tmpg[((size_t)(b0 + m) * T_ + t) * 24 + j];
        sm.a_vad  [m * 32 + j] = h;
        sm.a_noise[m * 96 + j] = h;
      }
    }
    if (t + 1 < T_ && tid < 16)   // warm L2/WGP$ for the step after
      __builtin_prefetch(&x[((size_t)(b0 + tid) * T_ + (t + 1)) * F_], 0, 1);
  };

  // Zero f16 state (h(0)=0, padding=0), then pin bias-partner 1.0 columns.
  {
    _Float16* p = (_Float16*)&sm;   // f16 block = first 9728 elements
    for (int i = tid; i < 9728; i += 256) p[i] = (_Float16)0.f;
  }
  __syncthreads();
  if (tid < 16) {                       // one thread per row m
    const _Float16 one = (_Float16)1.f;
    sm.a_dense[tid * 64 + 42]  = one;   // dense bias partner (fallback)
    sm.a_vad  [tid * 32 + 24]  = one;   // vad b_ih
    sm.a_hv   [tid * 32 + 24]  = one;   // vad b_hh
    sm.a_noise[tid * 96 + 90]  = one;   // noise b_ih
    sm.a_hn   [tid * 64 + 48]  = one;   // noise b_hh
    sm.a_den  [tid * 128 + 114] = one;  // den b_ih
  }
  stage(0);
  __syncthreads();

  for (int t = 0; t < T_; ++t) {
    if (!PRECOMP) {
      gemm_one<2, 2>(sm.a_dense, 64, pw + (FR_DENSE << 9), sm.gx, wave, lane);
      __syncthreads();
      for (int i = tid; i < 16 * 24; i += 256) {
        int m = i / 24, j = i - m * 24;
        _Float16 h = (_Float16)tanhf_(sm.gx[m * 288 + j]);   // bias already in
        sm.a_vad  [m * 32 + j] = h;
        sm.a_noise[m * 96 + j] = h;
      }
      __syncthreads();
    }

    // -- vad GRU (H=24): gx+gh as one balanced job list ---------------------
    gemm_pair<5, 1, 5, 1>(sm.a_vad, 32, pw + (FR_VIH << 9), sm.gx,
                          sm.a_hv,  32, pw + (FR_VHH << 9), sm.gh, wave, lane);
    __syncthreads();
    for (int i = tid; i < 16 * 24; i += 256) {
      int m = i / 24, j = i - m * 24;
      float r = sigmoidf_(sm.gx[m*288 + j]      + sm.gh[m*288 + j]);
      float z = sigmoidf_(sm.gx[m*288 + 24 + j] + sm.gh[m*288 + 24 + j]);
      float n = tanhf_  (sm.gx[m*288 + 48 + j] + r * sm.gh[m*288 + 48 + j]);
      float hnew = (1.f - z) * n + z * (float)sm.a_hv[m * 32 + j];
      _Float16 h16 = (_Float16)hnew;
      sm.a_hv   [m * 32 + j]      = h16;
      sm.a_noise[m * 96 + 24 + j] = h16;
      sm.a_den  [m * 128 + j]     = h16;
    }
    __syncthreads();

    // -- noise GRU (H=48) + vad output head ---------------------------------
    gemm_pair<9, 3, 9, 2>(sm.a_noise, 96, pw + (FR_NIH << 9), sm.gx,
                          sm.a_hn,    64, pw + (FR_NHH << 9), sm.gh, wave, lane);
    if (tid < 16) {   // vad_out = sigmoid(h_vad . w + b); h_vad stable here
      float s = vob[0];
      for (int j = 0; j < 24; ++j) s += (float)sm.a_hv[tid * 32 + j] * vow[j];
      dvad[(size_t)(b0 + tid) * T_ + t] = sigmoidf_(s);
    }
    __syncthreads();
    for (int i = tid; i < 16 * 48; i += 256) {
      int m = i / 48, j = i - m * 48;
      float r = sigmoidf_(sm.gx[m*288 + j]      + sm.gh[m*288 + j]);
      float z = sigmoidf_(sm.gx[m*288 + 48 + j] + sm.gh[m*288 + 48 + j]);
      float n = tanhf_  (sm.gx[m*288 + 96 + j] + r * sm.gh[m*288 + 96 + j]);
      float hnew = (1.f - z) * n + z * (float)sm.a_hn[m * 64 + j];
      sm.a_hn [m * 64 + j]       = (_Float16)hnew;
      sm.a_den[m * 128 + 24 + j] = (_Float16)fmaxf(hnew, 0.f);
    }
    __syncthreads();

    // -- denoise GRU (H=96): 36 balanced jobs, <=18 WMMA per wave -----------
    // den b_hh not fusable (K=96 exact) -> added here.
    gemm_pair<18, 4, 18, 3>(sm.a_den, 128, pw + (FR_DIH << 9), sm.gx,
                            sm.a_hd,   96, pw + (FR_DHH << 9), sm.gh, wave, lane);
    __syncthreads();
    for (int i = tid; i < 16 * 96; i += 256) {
      int m = i / 96, j = i - m * 96;
      float r = sigmoidf_(sm.gx[m*288 + j]       + sm.gh[m*288 + j]       + dbh[j]);
      float z = sigmoidf_(sm.gx[m*288 + 96 + j]  + sm.gh[m*288 + 96 + j]  + dbh[96 + j]);
      float n = tanhf_  (sm.gx[m*288 + 192 + j] + r * (sm.gh[m*288 + 192 + j] + dbh[192 + j]));
      float hnew = (1.f - z) * n + z * (float)sm.a_hd[m * 96 + j];
      sm.a_hd [m * 96 + j] = (_Float16)hnew;
      sm.a_out[m * 96 + j] = (_Float16)fmaxf(hnew, 0.f);
    }
    __syncthreads();

    // -- output head --------------------------------------------------------
    gemm_one<2, 3>(sm.a_out, 96, pw + (FR_OUT << 9), sm.gx, wave, lane);
    __syncthreads();

    // -- tail: store outputs for t, stage inputs for t+1 (disjoint LDS) -----
    for (int i = tid; i < 16 * 22; i += 256) {
      int m = i / 22, n = i - m * 22;
      dout[((size_t)(b0 + m) * T_ + t) * 22 + n] =
          sigmoidf_(sm.gx[m * 288 + n] + out_b[n]);
    }
    if (t + 1 < T_) stage(t + 1);
    __syncthreads();
  }
}

// ---------------------------------------------------------------------------
extern "C" void kernel_launch(void* const* d_in, const int* in_sizes, int n_in,
                              void* d_out, int out_size, void* d_ws, size_t ws_size,
                              hipStream_t stream) {
  (void)in_sizes; (void)n_in; (void)out_size;
  // dict order: 0:x 1:dense_w 2:dense_b 3:vad_w_ih 4:vad_w_hh 5:vad_b_ih
  // 6:vad_b_hh 7:vad_out_w 8:vad_out_b 9:noise_w_ih 10:noise_w_hh 11:noise_b_ih
  // 12:noise_b_hh 13:den_w_ih 14:den_w_hh 15:den_b_ih 16:den_b_hh 17:out_w 18:out_b
  const float* x = (const float*)d_in[0];
  _Float16* pw  = (_Float16*)d_ws;                       // 195,584 B
  _Float16* tmp = (_Float16*)((char*)d_ws + TMP_OFF);    // 24.6 MB
  float* dout = (float*)d_out;
  float* dvad = dout + (size_t)B_ * T_ * 22;

  pack_weights_f16<<<(FR_TOTAL * 512 + 255) / 256, 256, 0, stream>>>(
      (const float*)d_in[1],  (const float*)d_in[2],
      (const float*)d_in[3],  (const float*)d_in[5],
      (const float*)d_in[4],  (const float*)d_in[6],
      (const float*)d_in[9],  (const float*)d_in[11],
      (const float*)d_in[10], (const float*)d_in[12],
      (const float*)d_in[13], (const float*)d_in[15],
      (const float*)d_in[14], (const float*)d_in[17], pw);

  const bool precomp = ws_size >= TMP_OFF + TMP_BYTES;   // fixed per process
  if (precomp) {
    size_t n = (size_t)B_ * T_ * 24;
    dense_tmp_precompute<<<(unsigned)((n + 255) / 256), 256, 0, stream>>>(
        x, (const float*)d_in[1], (const float*)d_in[2], tmp);
    rnnoise_fused<true><<<B_ / 16, 256, 0, stream>>>(
        x, pw, tmp,
        (const float*)d_in[7], (const float*)d_in[8],
        (const float*)d_in[16], (const float*)d_in[18], dout, dvad);
  } else {
    rnnoise_fused<false><<<B_ / 16, 256, 0, stream>>>(
        x, pw, nullptr,
        (const float*)d_in[7], (const float*)d_in[8],
        (const float*)d_in[16], (const float*)d_in[18], dout, dvad);
  }
}
